// SwinTransformerBlock_61598420959490
// MI455X (gfx1250) — matrix operand order repaired and checked by reading, not compiled, not run on hardware
//
#include <hip/hip_runtime.h>
#include <hip/hip_bf16.h>
#include <math.h>

// Problem constants (Swin block: B=64, H=W=56, C=192, NH=6, WS=7, SS=3)
#define BATCH   64
#define IMH     56
#define IMW     56
#define CH      192
#define NHEAD   6
#define WSZ     7
#define SHIFT   3
#define NTOK    49          // WSZ*WSZ
#define NWIN    64          // (56/7)^2
#define HD      32          // CH/NHEAD
#define MLPH    768
#define LL      3136        // IMH*IMW
#define M1      200704      // BATCH*LL == BATCH*NWIN*NTOK  (== 1568*128)

typedef __attribute__((ext_vector_type(16))) __bf16        v16bf;
typedef __attribute__((ext_vector_type(16))) unsigned short v16us;
typedef __attribute__((ext_vector_type(8)))  unsigned short v8us;
typedef __attribute__((ext_vector_type(8)))  float          v8f;

static __device__ __forceinline__ unsigned short f2bf(float f) {
  unsigned int u = __float_as_uint(f);
  u += 0x7FFFu + ((u >> 16) & 1u);          // round-to-nearest-even
  return (unsigned short)(u >> 16);
}
static __device__ __forceinline__ float bf2f(unsigned short h) {
  return __uint_as_float(((unsigned int)h) << 16);
}

static __device__ __forceinline__ v8f wmma_bf16(v16us a, v16us b, v8f c) {
  return __builtin_amdgcn_wmma_f32_16x16x32_bf16(
      false, __builtin_bit_cast(v16bf, a),
      false, __builtin_bit_cast(v16bf, b),
      (short)0, c, false, false);
}

// A fragment: 16x32 bf16 tile at `base` (row-major, row stride = STRIDE ushorts).
// Lanes 0-15 row M=l16 hold K {hf*8..+7, 16+hf*8..+7} per ISA 7.12.2.
template <int STRIDE>
static __device__ __forceinline__ v16us load_a_frag(const unsigned short* base,
                                                    int l16, int hf) {
  const unsigned short* p = base + l16 * STRIDE + hf * 8;
  v8us lo = *(const v8us*)(p);
  v8us hi = *(const v8us*)(p + 16);
  return __builtin_shufflevector(lo, hi, 0,1,2,3,4,5,6,7,8,9,10,11,12,13,14,15);
}

// B fragment: B is K(32) x N(16); memory holds Bt[n][k] (i.e. weight rows / k^T
// rows), row stride = STRIDE. Lane column n=l16, K {hf*8..+7, 16+hf*8..+7}.
template <int STRIDE>
static __device__ __forceinline__ v16us load_b_frag(const unsigned short* base,
                                                    int l16, int hf) {
  const unsigned short* p = base + l16 * STRIDE + hf * 8;
  v8us lo = *(const v8us*)(p);
  v8us hi = *(const v8us*)(p + 16);
  return __builtin_shufflevector(lo, hi, 0,1,2,3,4,5,6,7,8,9,10,11,12,13,14,15);
}

// ---------------------------------------------------------------- weights cvt
__global__ __launch_bounds__(256) void f32bf16_kernel(
    const float* __restrict__ in, unsigned short* __restrict__ out, int n) {
  int i = blockIdx.x * 256 + threadIdx.x;
  if (i < n) out[i] = f2bf(in[i]);
}

// -------------------------------------------------------- LayerNorm (+window)
// One wave (32 lanes) per row of 192; WINDOW=true also does cyclic shift +
// window partition (row = ((b*NWIN + w)*NTOK + n)); else identity row map.
template <bool WINDOW>
__global__ __launch_bounds__(256) void ln_kernel(
    const float* __restrict__ xin, const float* __restrict__ gam,
    const float* __restrict__ bet, unsigned short* __restrict__ out) {
  const int row  = blockIdx.x * 8 + (threadIdx.x >> 5);
  const int lane = threadIdx.x & 31;
  long long src;
  if (WINDOW) {
    int n   = row % NTOK;
    int wiw = row / NTOK;
    int w   = wiw % NWIN;
    int b   = wiw / NWIN;
    int hh  = (w >> 3) * WSZ + n / WSZ;        // coords in rolled image
    int ww  = (w & 7)  * WSZ + n % WSZ;
    int hs  = hh + SHIFT; if (hs >= IMH) hs -= IMH;   // roll(-3) source
    int wsr = ww + SHIFT; if (wsr >= IMW) wsr -= IMW;
    src = ((long long)b * LL + hs * IMW + wsr) * CH;
  } else {
    src = (long long)row * CH;
  }
  const float* rp = xin + src;
  float v0[6];
  float s = 0.f, ss = 0.f;
#pragma unroll
  for (int i = 0; i < 6; ++i) {
    float v = rp[lane + 32 * i];
    v0[i] = v; s += v; ss += v * v;
  }
#pragma unroll
  for (int m = 16; m >= 1; m >>= 1) {
    s  += __shfl_xor(s,  m, 32);
    ss += __shfl_xor(ss, m, 32);
  }
  float mean = s  * (1.f / 192.f);
  float var  = ss * (1.f / 192.f) - mean * mean;
  float rstd = rsqrtf(var + 1e-5f);
  unsigned short* op = out + (long long)row * CH;
#pragma unroll
  for (int i = 0; i < 6; ++i) {
    int c = lane + 32 * i;
    op[c] = f2bf((v0[i] - mean) * rstd * gam[c] + bet[c]);
  }
}

// ------------------------------------------------------------- WMMA GEMM
// out[M,N] = A[M,K] @ W[N,K]^T, A/W bf16, acc f32. BM=128 BN=64 BK=32,
// 256 threads = 8 waves (4x2), each wave 2x2 WMMA tiles (32x32).
// MODE 0: +bias           -> bf16 (qkv)
// MODE 1: +bias, scatter (window-reverse + unshift) + shortcut -> f32 (proj)
// MODE 2: +bias, BN, GELU -> bf16 (fc1)
// MODE 3: +bias, +extra   -> f32 (fc2 / final out)
template <int MODE>
__global__ __launch_bounds__(256) void gemm_wmma_kernel(
    const unsigned short* __restrict__ A, const unsigned short* __restrict__ Wt,
    const float* __restrict__ bias, const float* __restrict__ extra,
    void* __restrict__ outp, int N, int K,
    const float* __restrict__ bng, const float* __restrict__ bnb) {
  __shared__ __align__(16) unsigned short As[128][32];
  __shared__ __align__(16) unsigned short Bs[64][32];
  const int tid  = threadIdx.x;
  const int lane = tid & 31;
  const int wv   = tid >> 5;
  const int hf   = lane >> 4;
  const int l16  = lane & 15;
  const int wm   = wv & 3;
  const int wn   = wv >> 2;
  const long long rowBase = (long long)blockIdx.y * 128;
  const int colBase = blockIdx.x * 64;

  v8f acc[2][2] = {};

  const int ar = tid >> 1, ako = (tid & 1) * 16;
  const int br = tid >> 2, bko = (tid & 3) * 8;
  const unsigned short* agp = A  + (rowBase + ar) * K + ako;
  const unsigned short* bgp = Wt + (long long)(colBase + br) * K + bko;

  for (int k0 = 0; k0 < K; k0 += 32) {
    *(v8us*)&As[ar][ako]     = *(const v8us*)(agp);
    *(v8us*)&As[ar][ako + 8] = *(const v8us*)(agp + 8);
    *(v8us*)&Bs[br][bko]     = *(const v8us*)(bgp);
    agp += 32; bgp += 32;
    if (k0 + 32 < K) {           // hint next K-slab -> global_prefetch_b8
      __builtin_prefetch(agp, 0, 1);
      __builtin_prefetch(bgp, 0, 1);
    }
    __syncthreads();
    v16us af0 = load_a_frag<32>(&As[wm * 32][0],      l16, hf);
    v16us af1 = load_a_frag<32>(&As[wm * 32 + 16][0], l16, hf);
    v16us bf0 = load_b_frag<32>(&Bs[wn * 32][0],      l16, hf);
    v16us bf1 = load_b_frag<32>(&Bs[wn * 32 + 16][0], l16, hf);
    acc[0][0] = wmma_bf16(af0, bf0, acc[0][0]);
    acc[0][1] = wmma_bf16(af0, bf1, acc[0][1]);
    acc[1][0] = wmma_bf16(af1, bf0, acc[1][0]);
    acc[1][1] = wmma_bf16(af1, bf1, acc[1][1]);
    __syncthreads();
  }

#pragma unroll
  for (int mt = 0; mt < 2; ++mt)
#pragma unroll
    for (int nt = 0; nt < 2; ++nt)
#pragma unroll
      for (int r = 0; r < 8; ++r) {
        long long gr = rowBase + wm * 32 + mt * 16 + hf * 8 + r;
        int gc = colBase + wn * 32 + nt * 16 + l16;
        float v = acc[mt][nt][r] + bias[gc];
        if (MODE == 0) {
          ((unsigned short*)outp)[gr * N + gc] = f2bf(v);
        } else if (MODE == 1) {
          int rr  = (int)gr;
          int n   = rr % NTOK;
          int wiw = rr / NTOK;
          int w   = wiw % NWIN, b = wiw / NWIN;
          int hh  = (w >> 3) * WSZ + n / WSZ;
          int ww2 = (w & 7)  * WSZ + n % WSZ;
          int ho  = hh + SHIFT;  if (ho >= IMH) ho -= IMH;   // roll(+3)
          int wo  = ww2 + SHIFT; if (wo >= IMW) wo -= IMW;
          long long oi = ((long long)b * LL + ho * IMW + wo) * CH + gc;
          ((float*)outp)[oi] = v + extra[oi];                 // + shortcut
        } else if (MODE == 2) {
          float z  = v * 0.9999950000375f * bng[gc] + bnb[gc]; // eval BN
          float ge = 0.5f * z * (1.0f + erff(z * 0.7071067811865475f));
          ((unsigned short*)outp)[gr * N + gc] = f2bf(ge);
        } else {
          long long oi = gr * N + gc;
          ((float*)outp)[oi] = v + extra[oi];                 // + residual
        }
      }
}

// ----------------------------------------------------------- fused attention
// One block (128 thr, 4 waves) per (window, head). S = q@k^T via WMMA (K=HD=32
// is one WMMA K-step), in-LDS softmax with computed rel-pos bias + shift mask,
// sigmoid gate, O = P@V via WMMA.
__global__ __launch_bounds__(128) void attn_kernel(
    const unsigned short* __restrict__ qkv, const float* __restrict__ rpb,
    const float* __restrict__ gate, unsigned short* __restrict__ obuf) {
  __shared__ __align__(16) unsigned short q_s[64][32];
  __shared__ __align__(16) unsigned short k_s[64][32];
  __shared__ __align__(16) unsigned short vT_s[32][64];
  __shared__ __align__(16) float          s_s[64][64];
  __shared__ __align__(16) unsigned short p_s[64][64];

  const int bh = blockIdx.x;
  const int h  = bh % NHEAD;
  const int wi = bh / NHEAD;
  const int w  = wi % NWIN;
  const int tid = threadIdx.x;

  for (int idx = tid; idx < 64 * 32; idx += 128) {
    int tok = idx >> 5, d = idx & 31;
    unsigned short qv = 0, kv = 0, vv = 0;
    if (tok < NTOK) {
      long long base = ((long long)wi * NTOK + tok) * 576 + h * HD + d;
      qv = qkv[base];
      kv = qkv[base + 192];
      vv = qkv[base + 384];
    }
    q_s[tok][d]  = qv;
    k_s[tok][d]  = kv;
    vT_s[d][tok] = vv;
  }
  __syncthreads();

  const int lane = tid & 31, wv = tid >> 5;
  const int hf = lane >> 4, l16 = lane & 15;

  // ---- S = q @ k^T (wave wv owns row-tile wv) -------------------------
  {
    v16us af = load_a_frag<32>(&q_s[wv * 16][0], l16, hf);
#pragma unroll
    for (int nt = 0; nt < 4; ++nt) {
      v16us bf = load_b_frag<32>(&k_s[nt * 16][0], l16, hf);
      v8f acc = {};
      acc = wmma_bf16(af, bf, acc);
#pragma unroll
      for (int r = 0; r < 8; ++r)
        s_s[wv * 16 + hf * 8 + r][nt * 16 + l16] = acc[r];
    }
  }
  __syncthreads();

  // ---- softmax (thread t owns row t) ----------------------------------
  if (tid < 64) {
    const int i  = tid;
    const int ii = (i < NTOK) ? i : (NTOK - 1);   // pad rows: don't-care, stay in-bounds
    const int ih = ii / WSZ, iw = ii % WSZ;
    const int hp = (w >> 3) * WSZ + ih;
    const int wp = (w & 7)  * WSZ + iw;
    const int gi = (hp < 49 ? 0 : (hp < 53 ? 1 : 2)) * 3 +
                   (wp < 49 ? 0 : (wp < 53 ? 1 : 2));
    const float scale = 0.17677669529663687f;     // 1/sqrt(32)
    float mx = -1e30f;
    for (int j = 0; j < NTOK; ++j) {
      int jh = j / WSZ, jw = j % WSZ;
      int rel = (ih - jh + 6) * 13 + (iw - jw + 6);
      int hq = (w >> 3) * WSZ + jh, wq = (w & 7) * WSZ + jw;
      int gj = (hq < 49 ? 0 : (hq < 53 ? 1 : 2)) * 3 +
               (wq < 49 ? 0 : (wq < 53 ? 1 : 2));
      float f = s_s[i][j] * scale + rpb[rel * NHEAD + h] +
                (gi != gj ? -100.f : 0.f);
      s_s[i][j] = f;
      mx = fmaxf(mx, f);
    }
    float sum = 0.f;
    for (int j = 0; j < NTOK; ++j) sum += __expf(s_s[i][j] - mx);
    const float gsig = 1.f / (1.f + __expf(-gate[h]));
    const float inv  = gsig / sum;                // post-softmax gate folded in
    for (int j = 0; j < NTOK; ++j) p_s[i][j] = f2bf(__expf(s_s[i][j] - mx) * inv);
    for (int j = NTOK; j < 64; ++j) p_s[i][j] = 0;
  }
  __syncthreads();

  // ---- O = P @ V ------------------------------------------------------
  {
    v8f acc0 = {}, acc1 = {};
#pragma unroll
    for (int kk = 0; kk < 2; ++kk) {
      v16us af  = load_a_frag<64>(&p_s[wv * 16][kk * 32], l16, hf);
      v16us bf0 = load_b_frag<64>(&vT_s[0][kk * 32],  l16, hf);
      v16us bf1 = load_b_frag<64>(&vT_s[16][kk * 32], l16, hf);
      acc0 = wmma_bf16(af, bf0, acc0);
      acc1 = wmma_bf16(af, bf1, acc1);
    }
#pragma unroll
    for (int r = 0; r < 8; ++r) {
      int i = wv * 16 + hf * 8 + r;
      if (i < NTOK) {
        long long ob = ((long long)wi * NTOK + i) * CH + h * HD;
        obuf[ob + l16]      = f2bf(acc0[r]);
        obuf[ob + 16 + l16] = f2bf(acc1[r]);
      }
    }
  }
}

// ---------------------------------------------------------------------------
extern "C" void kernel_launch(void* const* d_in, const int* in_sizes, int n_in,
                              void* d_out, int out_size, void* d_ws, size_t ws_size,
                              hipStream_t stream) {
  (void)in_sizes; (void)n_in; (void)out_size; (void)ws_size;
  const float* x      = (const float*)d_in[0];
  const float* n1g    = (const float*)d_in[1];
  const float* n1b    = (const float*)d_in[2];
  const float* qkv_w  = (const float*)d_in[3];
  const float* qkv_b  = (const float*)d_in[4];
  const float* rpb    = (const float*)d_in[5];
  const float* gate   = (const float*)d_in[6];
  const float* proj_w = (const float*)d_in[7];
  const float* proj_b = (const float*)d_in[8];
  const float* n2g    = (const float*)d_in[9];
  const float* n2b    = (const float*)d_in[10];
  const float* fc1_w  = (const float*)d_in[11];
  const float* fc1_b  = (const float*)d_in[12];
  const float* bn_g   = (const float*)d_in[13];
  const float* bn_b   = (const float*)d_in[14];
  const float* fc2_w  = (const float*)d_in[15];
  const float* fc2_b  = (const float*)d_in[16];

  char* ws = (char*)d_ws;
  size_t off = 0;
  unsigned short* w_qkv  = (unsigned short*)(ws + off); off += (size_t)576 * 192 * 2;
  unsigned short* w_proj = (unsigned short*)(ws + off); off += (size_t)192 * 192 * 2;
  unsigned short* w_fc1  = (unsigned short*)(ws + off); off += (size_t)768 * 192 * 2;
  unsigned short* w_fc2  = (unsigned short*)(ws + off); off += (size_t)192 * 768 * 2;
  off = (off + 255) & ~(size_t)255;

  // Big region: phase-1 holds yw (bf16) + qkv (bf16); phase-2 reused for FC1
  // hidden (bf16). Both layouts are exactly M1*768*2 bytes.
  size_t offBig = off;
  unsigned short* yw   = (unsigned short*)(ws + offBig);
  unsigned short* qkvb = (unsigned short*)(ws + offBig + (size_t)M1 * CH * 2);
  unsigned short* hbuf = (unsigned short*)(ws + offBig);
  size_t offO = offBig + (size_t)M1 * MLPH * 2;
  unsigned short* obuf = (unsigned short*)(ws + offO);   // attention out (bf16)
  unsigned short* zbuf = (unsigned short*)(ws + offO);   // LN2 out reuses it
  float* x1 = (float*)(ws + offO + (size_t)M1 * CH * 2); // post-attn residual

  // weights -> bf16
  f32bf16_kernel<<<(576 * 192 + 255) / 256, 256, 0, stream>>>(qkv_w,  w_qkv,  576 * 192);
  f32bf16_kernel<<<(192 * 192 + 255) / 256, 256, 0, stream>>>(proj_w, w_proj, 192 * 192);
  f32bf16_kernel<<<(768 * 192 + 255) / 256, 256, 0, stream>>>(fc1_w,  w_fc1,  768 * 192);
  f32bf16_kernel<<<(192 * 768 + 255) / 256, 256, 0, stream>>>(fc2_w,  w_fc2,  192 * 768);

  // LN1 + cyclic shift + window partition
  ln_kernel<true><<<M1 / 8, 256, 0, stream>>>(x, n1g, n1b, yw);

  // QKV projection: [M1,192] @ [576,192]^T
  gemm_wmma_kernel<0><<<dim3(576 / 64, M1 / 128), 256, 0, stream>>>(
      yw, w_qkv, qkv_b, nullptr, qkvb, 576, 192, nullptr, nullptr);

  // windowed attention (one block per window-head)
  attn_kernel<<<BATCH * NWIN * NHEAD, 128, 0, stream>>>(qkvb, rpb, gate, obuf);

  // output projection + window reverse + unshift + shortcut
  gemm_wmma_kernel<1><<<dim3(192 / 64, M1 / 128), 256, 0, stream>>>(
      obuf, w_proj, proj_b, x, x1, 192, 192, nullptr, nullptr);

  // LN2
  ln_kernel<false><<<M1 / 8, 256, 0, stream>>>(x1, n2g, n2b, zbuf);

  // FC1 + BN + GELU
  gemm_wmma_kernel<2><<<dim3(768 / 64, M1 / 128), 256, 0, stream>>>(
      zbuf, w_fc1, fc1_b, nullptr, hbuf, 768, 192, bn_g, bn_b);

  // FC2 + residual -> final output (f32)
  gemm_wmma_kernel<3><<<dim3(192 / 64, M1 / 128), 256, 0, stream>>>(
      hbuf, w_fc2, fc2_b, x1, d_out, 192, 768, nullptr, nullptr);
}
